// RN_41893111005305
// MI455X (gfx1250) — matrix operand order repaired and checked by reading, not compiled, not run on hardware
//
#include <hip/hip_runtime.h>
#include <hip/hip_bf16.h>

typedef _Float16 v16h __attribute__((ext_vector_type(16)));
typedef _Float16 v8h  __attribute__((ext_vector_type(8)));
typedef float    v8f  __attribute__((ext_vector_type(8)));
typedef float    v4f  __attribute__((ext_vector_type(4)));

#define BATCH 32
#define CCH   24
#define DD    12
#define NOBJ  144            // 12*12 objects
#define KCH   26             // channels after coord concat
#define QDIM  11
#define NH    256
#define JB    (NOBJ/16)      // 9 j-blocks per i
#define NTILES (NOBJ*JB)     // 1296 tiles of 16 pairs (fixed i, 16 consecutive j)
#define NSPLIT 18
#define TILES_PER_SPLIT (NTILES/NSPLIT)  // 72

// ---------------------------------------------------------------------------
// Phase A1: qb[b,:] = question[b] @ Wg1[52:63] + bg1 ; zero the s accumulator
// ---------------------------------------------------------------------------
__global__ void rn_prep_q(const float* __restrict__ question,
                          const float* __restrict__ Wg1,
                          const float* __restrict__ bg1,
                          float* __restrict__ qb,
                          float* __restrict__ s_acc) {
  int b = blockIdx.x;
  int h = threadIdx.x;
  float acc = bg1[h];
#pragma unroll
  for (int q = 0; q < QDIM; ++q)
    acc = fmaf(question[b * QDIM + q], Wg1[(2 * KCH + q) * NH + h], acc);
  qb[b * NH + h]    = acc;
  s_acc[b * NH + h] = 0.f;
}

// ---------------------------------------------------------------------------
// Phase A2: xA[b,o,:] = x[b,o,:] @ Wg1[0:26]
//           xBq[b,o,:] = x[b,o,:] @ Wg1[26:52] + qb[b,:]   (qb pre-folded)
// x = image channels + coord channels (cx = col/11, cy = row/11)
// ---------------------------------------------------------------------------
__global__ void rn_prep_obj(const float* __restrict__ image,
                            const float* __restrict__ Wg1,
                            const float* __restrict__ qb,
                            float* __restrict__ xA,
                            float* __restrict__ xBq) {
  int bo = blockIdx.x;              // 0 .. 32*144-1
  int b  = bo / NOBJ;
  int o  = bo - b * NOBJ;
  int y  = o / DD;
  int xx = o - y * DD;

  __shared__ float xv[KCH];
  int t = threadIdx.x;
  if (t < KCH) {
    float v;
    if (t < CCH)        v = image[((b * CCH + t) * DD + y) * DD + xx];
    else if (t == CCH)  v = (float)xx * (1.0f / (DD - 1));
    else                v = (float)y  * (1.0f / (DD - 1));
    xv[t] = v;
  }
  __syncthreads();

  int h = t;                        // 256 threads, one output channel each
  float sa = 0.f, sb = 0.f;
#pragma unroll
  for (int c = 0; c < KCH; ++c) {
    float xc = xv[c];
    sa = fmaf(xc, Wg1[c * NH + h],          sa);
    sb = fmaf(xc, Wg1[(KCH + c) * NH + h],  sb);
  }
  xA [(b * NOBJ + o) * NH + h] = sa;
  xBq[(b * NOBJ + o) * NH + h] = sb + qb[b * NH + h];
}

// ---------------------------------------------------------------------------
// Main: tile = (i, 16 consecutive j). h1 row r = relu(xA[j0+r] + xBq[i]),
// converted to f16 into a double-buffered LDS tile. 16 waves each own one
// 16-col N-tile with Wg2 fragments resident in VGPRs; 8x v_wmma per tile
// (2 accumulators), relu(+bg2) row-sums folded, one f32 atomic per lane.
// ---------------------------------------------------------------------------
__global__ void __launch_bounds__(512, 1)
rn_pair_gemm(const float* __restrict__ xA,
             const float* __restrict__ xBq,
             const float* __restrict__ Wg2,
             const float* __restrict__ bg2,
             float* __restrict__ s_acc) {
  const int b     = blockIdx.x;
  const int split = blockIdx.y;
  const int tid   = threadIdx.x;
  const int lane  = tid & 31;
  const int wave  = tid >> 5;
  const int n0    = wave * 16;             // this wave's N-tile base
  const int col   = n0 + (lane & 15);      // output column for this lane
  const int hi    = lane >> 4;             // lane half (VGPR layout select)

  __shared__ __align__(16) _Float16 h1[2][16 * NH];   // 2 x 8 KB tiles

  // --- Preload B fragments of Wg2 (f32->f16), WMMA B layout, all 8 K-chunks.
  // 16-bit B 32x16: lanes 0-15 hold K=0..7,16..23 (N=lane&15), lanes 16-31 K=8..15,24..31.
  v16h bf[8];
#pragma unroll
  for (int kc = 0; kc < 8; ++kc) {
#pragma unroll
    for (int t2 = 0; t2 < 16; ++t2) {
      int k = kc * 32 + (t2 >> 3) * 16 + hi * 8 + (t2 & 7);
      bf[kc][t2] = (_Float16)Wg2[k * NH + col];
    }
  }
  const float bg2v = bg2[col];

  const float* __restrict__ xAb = xA  + b * NOBJ * NH;
  const float* __restrict__ xBb = xBq + b * NOBJ * NH;

  // Fill mapping: thread owns row r = tid>>5 (0..15), 8 consecutive h at c0.
  const int fr  = tid >> 5;
  const int fc0 = (tid & 31) * 8;

  // Load the two 8-float global runs for tile's row fr (issued early).
  auto load_tile = [&](int tile, v4f& a0, v4f& a1, v4f& b0, v4f& b1) {
    int i  = tile / JB;
    int jb = tile - i * JB;
    const float* pa = xAb + (jb * 16 + fr) * NH + fc0;
    const float* pb = xBb + i * NH + fc0;
    a0 = *(const v4f*)(pa);
    a1 = *(const v4f*)(pa + 4);
    b0 = *(const v4f*)(pb);
    b1 = *(const v4f*)(pb + 4);
  };
  // relu + f32->f16 + single ds_store_b128 (issued late, after A-frag loads).
  auto store_tile = [&](int buf, v4f a0, v4f a1, v4f b0, v4f b1) {
    v8h o;
#pragma unroll
    for (int t2 = 0; t2 < 4; ++t2) {
      o[t2]     = (_Float16)fmaxf(a0[t2] + b0[t2], 0.f);
      o[t2 + 4] = (_Float16)fmaxf(a1[t2] + b1[t2], 0.f);
    }
    *(v8h*)(&h1[buf][fr * NH + fc0]) = o;
  };

  const int tile0 = split * TILES_PER_SPLIT;
  float rowacc = 0.f;

  // Prologue: fill buffer 0.
  {
    v4f a0, a1, b0, b1;
    load_tile(tile0, a0, a1, b0, b1);
    store_tile(0, a0, a1, b0, b1);
  }
  __syncthreads();

  const _Float16* __restrict__ rowbase0 = &h1[0][(lane & 15) * NH];
  const _Float16* __restrict__ rowbase1 = &h1[1][(lane & 15) * NH];

  for (int t = 0; t < TILES_PER_SPLIT; ++t) {
    const int buf = t & 1;
    const bool more = (t + 1) < TILES_PER_SPLIT;

    // Issue next tile's global loads before the WMMA chain (latency hiding).
    v4f a0, a1, b0, b1;
    if (more) load_tile(tile0 + t + 1, a0, a1, b0, b1);

    // 8 WMMAs over K=256, two accumulators to break the RAW chain.
    v8f c0 = {}, c1 = {};
    const _Float16* __restrict__ row = buf ? rowbase1 : rowbase0;
#pragma unroll
    for (int kc = 0; kc < 8; kc += 2) {
      int k0 = kc * 32 + hi * 8;
      v8h r1 = *(const v8h*)(row + k0);              // ds_load_b128
      v8h r2 = *(const v8h*)(row + k0 + 16);         // ds_load_b128
      v8h r3 = *(const v8h*)(row + k0 + 32);
      v8h r4 = *(const v8h*)(row + k0 + 48);
      v16h aa, ab;
#pragma unroll
      for (int t2 = 0; t2 < 8; ++t2) {
        aa[t2] = r1[t2]; aa[t2 + 8] = r2[t2];
        ab[t2] = r3[t2]; ab[t2 + 8] = r4[t2];
      }
      c0 = __builtin_amdgcn_wmma_f32_16x16x32_f16(
               false, aa, false, bf[kc],     (short)0, c0, false, false);
      c1 = __builtin_amdgcn_wmma_f32_16x16x32_f16(
               false, ab, false, bf[kc + 1], (short)0, c1, false, false);
    }

    // Stage next tile into the other LDS buffer (after A-frag loads issued).
    if (more) store_tile(buf ^ 1, a0, a1, b0, b1);

    // relu(+bias) fold of the 8 rows this lane holds.
#pragma unroll
    for (int r = 0; r < 8; ++r)
      rowacc += fmaxf(c0[r] + c1[r] + bg2v, 0.f);

    __syncthreads();   // next buffer filled; current buffer free for reuse
  }

  // lanes L and L+16 hold the same column -> native f32 atomic handles both
  unsafeAtomicAdd(&s_acc[b * NH + col], rowacc);
}

// ---------------------------------------------------------------------------
// Final head: out = relu(s @ Wf1 + bf1) @ Wf2 + bf2
// ---------------------------------------------------------------------------
__global__ void rn_final(const float* __restrict__ s_acc,
                         const float* __restrict__ Wf1,
                         const float* __restrict__ bf1,
                         const float* __restrict__ Wf2,
                         const float* __restrict__ bf2,
                         float* __restrict__ out) {
  int b = blockIdx.x;
  int h = threadIdx.x;
  __shared__ float tbuf[NH];
  float acc = bf1[h];
  for (int k = 0; k < NH; ++k)
    acc = fmaf(s_acc[b * NH + k], Wf1[k * NH + h], acc);
  tbuf[h] = fmaxf(acc, 0.f);
  __syncthreads();
  if (h < 10) {
    float o = bf2[h];
    for (int k = 0; k < NH; ++k)
      o = fmaf(tbuf[k], Wf2[k * 10 + h], o);
    out[b * 10 + h] = o;
  }
}

// ---------------------------------------------------------------------------
extern "C" void kernel_launch(void* const* d_in, const int* in_sizes, int n_in,
                              void* d_out, int out_size, void* d_ws, size_t ws_size,
                              hipStream_t stream) {
  const float* image    = (const float*)d_in[0];
  const float* question = (const float*)d_in[1];
  const float* Wg1      = (const float*)d_in[2];
  const float* bg1      = (const float*)d_in[3];
  const float* Wg2      = (const float*)d_in[4];
  const float* bg2      = (const float*)d_in[5];
  const float* Wf1      = (const float*)d_in[6];
  const float* bf1      = (const float*)d_in[7];
  const float* Wf2      = (const float*)d_in[8];
  const float* bf2      = (const float*)d_in[9];
  float* out = (float*)d_out;

  float* ws    = (float*)d_ws;
  float* xA    = ws;                              // 32*144*256
  float* xBq   = xA + BATCH * NOBJ * NH;          // 32*144*256 (qb folded in)
  float* qb    = xBq + BATCH * NOBJ * NH;         // 32*256
  float* s_acc = qb + BATCH * NH;                 // 32*256

  rn_prep_q  <<<dim3(BATCH),        dim3(256), 0, stream>>>(question, Wg1, bg1, qb, s_acc);
  rn_prep_obj<<<dim3(BATCH * NOBJ), dim3(256), 0, stream>>>(image, Wg1, qb, xA, xBq);
  rn_pair_gemm<<<dim3(BATCH, NSPLIT), dim3(512), 0, stream>>>(xA, xBq, Wg2, bg2, s_acc);
  rn_final   <<<dim3(BATCH),        dim3(256), 0, stream>>>(s_acc, Wf1, bf1, Wf2, bf2, out);
}
